// Quantizer_49203145343432
// MI455X (gfx1250) — matrix operand order repaired and checked by reading, not compiled, not run on hardware
//
#include <hip/hip_runtime.h>

typedef __bf16  v16bf __attribute__((ext_vector_type(16)));
typedef float   v8f   __attribute__((ext_vector_type(8)));

#define N_TOK          16384      // B*H*W = 16*32*32
#define VOCAB          8192
#define KDIM           256
#define HWSZ           1024       // H*W
#define ROWS_PER_BLOCK 128

// fallback-kernel LDS stride (ushorts)
#define LDS_STRIDE_FB  272        // 256 + 16 pad -> 544B rows
// async-kernel tile geometry (ushorts): 16 rows x 264 (528B rows, 16B aligned)
#define TROW_US        264
#define MAT_US         (16 * TROW_US)      // 4224 ushorts = 8448 B per matrix
#define BUF_US         (2 * MAT_US)        // hi + lo      = 16896 B per buffer

__device__ __forceinline__ unsigned short f2bf(float f) {
  union { float f; unsigned u; } c; c.f = f;
  unsigned u = c.u;
  return (unsigned short)((u + 0x7FFFu + ((u >> 16) & 1u)) >> 16);  // RNE
}
__device__ __forceinline__ float bf2f(unsigned short s) {
  union { unsigned u; float f; } c; c.u = ((unsigned)s) << 16;
  return c.f;
}
__device__ __forceinline__ __bf16 bfbits(unsigned short s) {
  union { unsigned short s; __bf16 b; } c; c.s = s; return c.b;
}

// ---------------------------------------------------------------------------
// Prep (fast path): one wave per codebook row. Emits bf16 hi/lo split rows
// plus bias[v] = -0.5*||c_v||^2.
// ---------------------------------------------------------------------------
__global__ void vq_prep_kernel(const float* __restrict__ cb,
                               float* __restrict__ bias,
                               unsigned short* __restrict__ cbhi,
                               unsigned short* __restrict__ cblo) {
  int row  = (blockIdx.x * blockDim.x + threadIdx.x) >> 5;
  int lane = threadIdx.x & 31;
  if (row >= VOCAB) return;
  const float4* src = (const float4*)(cb + row * KDIM + lane * 8);
  float v[8];
  float4 a = src[0], b = src[1];
  v[0]=a.x; v[1]=a.y; v[2]=a.z; v[3]=a.w;
  v[4]=b.x; v[5]=b.y; v[6]=b.z; v[7]=b.w;
  unsigned short hs[8], ls[8];
  float s = 0.f;
#pragma unroll
  for (int i = 0; i < 8; ++i) {
    s += v[i] * v[i];
    unsigned short h = f2bf(v[i]);
    hs[i] = h;
    ls[i] = f2bf(v[i] - bf2f(h));
  }
  uint4 uh, ul;
  uh.x = (unsigned)hs[0] | ((unsigned)hs[1] << 16);
  uh.y = (unsigned)hs[2] | ((unsigned)hs[3] << 16);
  uh.z = (unsigned)hs[4] | ((unsigned)hs[5] << 16);
  uh.w = (unsigned)hs[6] | ((unsigned)hs[7] << 16);
  ul.x = (unsigned)ls[0] | ((unsigned)ls[1] << 16);
  ul.y = (unsigned)ls[2] | ((unsigned)ls[3] << 16);
  ul.z = (unsigned)ls[4] | ((unsigned)ls[5] << 16);
  ul.w = (unsigned)ls[6] | ((unsigned)ls[7] << 16);
  *(uint4*)(cbhi + row * KDIM + lane * 8) = uh;
  *(uint4*)(cblo + row * KDIM + lane * 8) = ul;
#pragma unroll
  for (int m = 16; m >= 1; m >>= 1) s += __shfl_xor(s, m, 32);
  if (lane == 0) bias[row] = -0.5f * s;
}

// bias-only prep (fallback path)
__global__ void vq_bias_kernel(const float* __restrict__ cb,
                               float* __restrict__ bias) {
  int wave = (blockIdx.x * blockDim.x + threadIdx.x) >> 5;
  int lane = threadIdx.x & 31;
  if (wave >= VOCAB) return;
  const float4* row = (const float4*)(cb + wave * KDIM + lane * 8);
  float s = 0.f;
#pragma unroll
  for (int g = 0; g < 2; ++g) {
    float4 x = row[g];
    s += x.x * x.x + x.y * x.y + x.z * x.z + x.w * x.w;
  }
#pragma unroll
  for (int m = 16; m >= 1; m >>= 1) s += __shfl_xor(s, m, 32);
  if (lane == 0) bias[wave] = -0.5f * s;
}

// ---------------------------------------------------------------------------
// Shared helper: load this wave's 16 ze rows (K=256) as bf16 hi/lo A fragments.
// A 16x32 layout (matches B): lanes 0-15 hold K 0..15, lanes 16-31 hold K 16..31
// ---------------------------------------------------------------------------
__device__ __forceinline__ void load_a_frags(const float* __restrict__ ze,
                                             int n0, int w, int lane,
                                             v16bf ah[8], v16bf al[8]) {
  const int m    = lane & 15;
  const int nrow = n0 + 16 * w + m;
  const int bb   = nrow >> 10;
  const int hw   = nrow & (HWSZ - 1);
  const float* zebase = ze + (bb << 18) + hw;
#pragma unroll
  for (int kc = 0; kc < 8; ++kc) {
    const int c0 = kc * 32 + ((lane & 16) ? 16 : 0);
#pragma unroll
    for (int j = 0; j < 8; ++j) {
      float x = zebase[(c0 + j) << 10];
      unsigned short h = f2bf(x);
      ah[kc][j] = bfbits(h);
      al[kc][j] = bfbits(f2bf(x - bf2f(h)));
    }
#pragma unroll
    for (int j = 0; j < 8; ++j) {
      float x = zebase[(c0 + 8 + j) << 10];
      unsigned short h = f2bf(x);
      ah[kc][8 + j] = bfbits(h);
      al[kc][8 + j] = bfbits(f2bf(x - bf2f(h)));
    }
  }
}

// ---------------------------------------------------------------------------
// Main kernel, fast path: async double-buffered codebook tiles (bf16 hi/lo
// precomputed in ws) streamed into LDS with global_load_async_to_lds_b128.
// B fragments are additionally double-buffered in registers one k-group ahead
// so WMMAs never wait on a just-issued ds_load (avoid s_wait_dscnt 0x0 drains).
// ---------------------------------------------------------------------------
__global__ void __launch_bounds__(256)
vq_main_async(const float* __restrict__ ze, const float* __restrict__ cb,
              const unsigned short* __restrict__ cbhi,
              const unsigned short* __restrict__ cblo,
              const float* __restrict__ bias, float* __restrict__ loss_part,
              float* __restrict__ zq, float* __restrict__ idx_out) {
  __shared__ __align__(16) unsigned short smem[2 * BUF_US];  // 33792 B
  __shared__ int   idxsm[ROWS_PER_BLOCK];
  __shared__ float red[256];

  const int tid  = threadIdx.x;
  const int lane = tid & 31;
  const int w    = tid >> 5;
  const int m    = lane & 15;
  const int n0   = blockIdx.x * ROWS_PER_BLOCK;

  v16bf ah[8], al[8];
  load_a_frags(ze, n0, w, lane, ah, al);

  float bv[8];
  int   bi[8];
#pragma unroll
  for (int i = 0; i < 8; ++i) { bv[i] = -3.4e38f; bi[i] = 0; }

  // ---- async-fill assignment: waves 0-3 move the hi matrix, 4-7 the lo
  // matrix; wave covers tile rows 4*wp..4*wp+3 (one b128 chunk per lane).
  const int wq = w >> 2;      // 0=hi matrix, 1=lo matrix
  const int wp = w & 3;
  const unsigned short* gsrc = wq ? cblo : cbhi;
  const unsigned short* gthr = gsrc + wp * 4 * KDIM + lane * 8;  // tile row 4*wp, 16B chunk
  const unsigned smem_off = (unsigned)(uintptr_t)(&smem[0]);     // flat addr[31:0] == LDS offset
  const unsigned lbase    = smem_off + wq * (MAT_US * 2) + wp * 4 * (TROW_US * 2) + lane * 16;

  // prologue: tile 0 -> buffer 0
#pragma unroll
  for (int j = 0; j < 4; ++j) {
    unsigned lds = lbase + j * (TROW_US * 2);
    unsigned long long ga = (unsigned long long)(uintptr_t)(gthr + j * KDIM);
    asm volatile("global_load_async_to_lds_b128 %0, %1, off"
                 :: "v"(lds), "v"(ga) : "memory");
  }

  for (int it = 0; it < VOCAB / 16; ++it) {
    const int vt  = it << 4;
    const int buf = it & 1;

    if (it + 1 < VOCAB / 16) {
      const unsigned lnext = lbase + ((it + 1) & 1) * (BUF_US * 2);
      const unsigned short* gnext = gthr + (it + 1) * 16 * KDIM;
#pragma unroll
      for (int j = 0; j < 4; ++j) {
        unsigned lds = lnext + j * (TROW_US * 2);
        unsigned long long ga = (unsigned long long)(uintptr_t)(gnext + j * KDIM);
        asm volatile("global_load_async_to_lds_b128 %0, %1, off"
                     :: "v"(lds), "v"(ga) : "memory");
      }
      asm volatile("s_wait_asynccnt 0x4" ::: "memory");  // tile `it` complete
    } else {
      asm volatile("s_wait_asynccnt 0x0" ::: "memory");
    }
    __syncthreads();

    // ---- one 16x16 subtile; B layout: lanes 0-15 K=0..15, lanes 16-31 K=16..31
    const unsigned short* bh_base =
        smem + buf * BUF_US + m * TROW_US + ((lane & 16) ? 16 : 0);
    const unsigned short* bl_base = bh_base + MAT_US;

    union Frag { uint4 q[2]; v16bf v; };
    Frag fh[2], fl[2];
    {
      const uint4* ph = (const uint4*)(bh_base);
      const uint4* pl = (const uint4*)(bl_base);
      fh[0].q[0] = ph[0]; fh[0].q[1] = ph[1];
      fl[0].q[0] = pl[0]; fl[0].q[1] = pl[1];
    }

    v8f acc0 = {0.f, 0.f, 0.f, 0.f, 0.f, 0.f, 0.f, 0.f};
    v8f acc1 = {0.f, 0.f, 0.f, 0.f, 0.f, 0.f, 0.f, 0.f};
#pragma unroll
    for (int kc = 0; kc < 8; ++kc) {
      const int cur = kc & 1;
      const int nxt = cur ^ 1;
      if (kc < 7) {  // prefetch next k-group's fragments during this group's WMMAs
        const uint4* ph = (const uint4*)(bh_base + (kc + 1) * 32);
        const uint4* pl = (const uint4*)(bl_base + (kc + 1) * 32);
        fh[nxt].q[0] = ph[0]; fh[nxt].q[1] = ph[1];
        fl[nxt].q[0] = pl[0]; fl[nxt].q[1] = pl[1];
      }
      if (kc & 1) {  // two independent accumulation chains for the matrix pipe
        acc1 = __builtin_amdgcn_wmma_f32_16x16x32_bf16(false, ah[kc], false, fh[cur].v,
                                                       (short)0, acc1, false, false);
        acc1 = __builtin_amdgcn_wmma_f32_16x16x32_bf16(false, al[kc], false, fh[cur].v,
                                                       (short)0, acc1, false, false);
        acc1 = __builtin_amdgcn_wmma_f32_16x16x32_bf16(false, ah[kc], false, fl[cur].v,
                                                       (short)0, acc1, false, false);
      } else {
        acc0 = __builtin_amdgcn_wmma_f32_16x16x32_bf16(false, ah[kc], false, fh[cur].v,
                                                       (short)0, acc0, false, false);
        acc0 = __builtin_amdgcn_wmma_f32_16x16x32_bf16(false, al[kc], false, fh[cur].v,
                                                       (short)0, acc0, false, false);
        acc0 = __builtin_amdgcn_wmma_f32_16x16x32_bf16(false, ah[kc], false, fl[cur].v,
                                                       (short)0, acc0, false, false);
      }
    }
    const int   vcol = vt + m;
    const float bia  = bias[vcol];
#pragma unroll
    for (int i = 0; i < 8; ++i) {
      float s = acc0[i] + acc1[i] + bia;
      if (s > bv[i]) { bv[i] = s; bi[i] = vcol; }   // strict > keeps smallest v
    }
    __syncthreads();
  }

  // ---- per-row argmax across the 16 lanes of each half (C/D: vgpr i -> M=i / i+8)
#pragma unroll
  for (int i = 0; i < 8; ++i) {
    float v = bv[i];
    int  ix = bi[i];
#pragma unroll
    for (int msk = 1; msk < 16; msk <<= 1) {
      float ov = __shfl_xor(v, msk, 32);
      int   oi = __shfl_xor(ix, msk, 32);
      if (ov > v || (ov == v && oi < ix)) { v = ov; ix = oi; }
    }
    if ((lane & 15) == 0) {
      int row = 16 * w + i + ((lane & 16) ? 8 : 0);
      idxsm[row] = ix;
      idx_out[n0 + row] = (float)ix;
    }
  }
  __syncthreads();

  // ---- epilogue: zq = codebook[idx] (== straight-through output), loss partial
  float lsum = 0.f;
  if (tid < ROWS_PER_BLOCK) {
    const int n   = n0 + tid;
    const int b2  = n >> 10;
    const int hw2 = n & (HWSZ - 1);
    const float* zb   = ze + (b2 << 18) + hw2;
    float*       qb   = zq + (b2 << 18) + hw2;
    const float* crow = cb + idxsm[tid] * KDIM;
#pragma unroll 4
    for (int c = 0; c < KDIM; ++c) {
      float q = crow[c];
      float x = zb[c << 10];
      float d = x - q;
      lsum += d * d;
      qb[c << 10] = q;
    }
  }
  red[tid] = lsum;
  __syncthreads();
  for (int s = 128; s > 0; s >>= 1) {
    if (tid < s) red[tid] += red[tid + s];
    __syncthreads();
  }
  if (tid == 0) loss_part[blockIdx.x] = red[0];
}

// ---------------------------------------------------------------------------
// Main kernel, fallback (no big workspace): converts codebook tiles in-kernel.
// ---------------------------------------------------------------------------
__global__ void __launch_bounds__(256)
vq_main_kernel(const float* __restrict__ ze, const float* __restrict__ cb,
               const float* __restrict__ bias, float* __restrict__ loss_part,
               float* __restrict__ zq, float* __restrict__ idx_out) {
  __shared__ __align__(16) unsigned short shi[32 * LDS_STRIDE_FB];
  __shared__ __align__(16) unsigned short slo[32 * LDS_STRIDE_FB];
  __shared__ int   idxsm[ROWS_PER_BLOCK];
  __shared__ float red[256];

  const int tid  = threadIdx.x;
  const int lane = tid & 31;
  const int w    = tid >> 5;
  const int m    = lane & 15;
  const int n0   = blockIdx.x * ROWS_PER_BLOCK;

  v16bf ah[8], al[8];
  load_a_frags(ze, n0, w, lane, ah, al);

  float bv[8];
  int   bi[8];
#pragma unroll
  for (int i = 0; i < 8; ++i) { bv[i] = -3.4e38f; bi[i] = 0; }

  const int fr = tid >> 3;
  const int fc = (tid & 7) * 32;

  for (int vt = 0; vt < VOCAB; vt += 32) {
    __syncthreads();
    {
      const float4* s4 = (const float4*)(cb + (vt + fr) * KDIM + fc);
      float v[32];
#pragma unroll
      for (int g = 0; g < 8; ++g) {
        float4 x = s4[g];
        v[4 * g + 0] = x.x; v[4 * g + 1] = x.y;
        v[4 * g + 2] = x.z; v[4 * g + 3] = x.w;
      }
      unsigned short hs[32], ls[32];
#pragma unroll
      for (int i = 0; i < 32; ++i) {
        unsigned short h = f2bf(v[i]);
        hs[i] = h;
        ls[i] = f2bf(v[i] - bf2f(h));
      }
      unsigned short* dh = shi + fr * LDS_STRIDE_FB + fc;
      unsigned short* dl = slo + fr * LDS_STRIDE_FB + fc;
#pragma unroll
      for (int q = 0; q < 4; ++q) {
        uint4 uh, ul;
        uh.x = (unsigned)hs[8*q+0] | ((unsigned)hs[8*q+1] << 16);
        uh.y = (unsigned)hs[8*q+2] | ((unsigned)hs[8*q+3] << 16);
        uh.z = (unsigned)hs[8*q+4] | ((unsigned)hs[8*q+5] << 16);
        uh.w = (unsigned)hs[8*q+6] | ((unsigned)hs[8*q+7] << 16);
        ul.x = (unsigned)ls[8*q+0] | ((unsigned)ls[8*q+1] << 16);
        ul.y = (unsigned)ls[8*q+2] | ((unsigned)ls[8*q+3] << 16);
        ul.z = (unsigned)ls[8*q+4] | ((unsigned)ls[8*q+5] << 16);
        ul.w = (unsigned)ls[8*q+6] | ((unsigned)ls[8*q+7] << 16);
        *(uint4*)(dh + 8 * q) = uh;
        *(uint4*)(dl + 8 * q) = ul;
      }
    }
    __syncthreads();

#pragma unroll
    for (int sub = 0; sub < 2; ++sub) {
      const unsigned short* bh_base =
          shi + (16 * sub + m) * LDS_STRIDE_FB + ((lane & 16) ? 16 : 0);
      const unsigned short* bl_base =
          slo + (16 * sub + m) * LDS_STRIDE_FB + ((lane & 16) ? 16 : 0);
      v8f acc = {0.f, 0.f, 0.f, 0.f, 0.f, 0.f, 0.f, 0.f};
#pragma unroll
      for (int kc = 0; kc < 8; ++kc) {
        union { uint4 q[2]; v16bf v; } Bh, Bl;
        const uint4* ph = (const uint4*)(bh_base + kc * 32);
        const uint4* pl = (const uint4*)(bl_base + kc * 32);
        Bh.q[0] = ph[0]; Bh.q[1] = ph[1];
        Bl.q[0] = pl[0]; Bl.q[1] = pl[1];
        acc = __builtin_amdgcn_wmma_f32_16x16x32_bf16(false, ah[kc], false, Bh.v,
                                                      (short)0, acc, false, false);
        acc = __builtin_amdgcn_wmma_f32_16x16x32_bf16(false, al[kc], false, Bh.v,
                                                      (short)0, acc, false, false);
        acc = __builtin_amdgcn_wmma_f32_16x16x32_bf16(false, ah[kc], false, Bl.v,
                                                      (short)0, acc, false, false);
      }
      const int   vcol = vt + 16 * sub + m;
      const float bia  = bias[vcol];
#pragma unroll
      for (int i = 0; i < 8; ++i) {
        float s = acc[i] + bia;
        if (s > bv[i]) { bv[i] = s; bi[i] = vcol; }
      }
    }
  }

#pragma unroll
  for (int i = 0; i < 8; ++i) {
    float v = bv[i];
    int  ix = bi[i];
#pragma unroll
    for (int msk = 1; msk < 16; msk <<= 1) {
      float ov = __shfl_xor(v, msk, 32);
      int   oi = __shfl_xor(ix, msk, 32);
      if (ov > v || (ov == v && oi < ix)) { v = ov; ix = oi; }
    }
    if ((lane & 15) == 0) {
      int row = 16 * w + i + ((lane & 16) ? 8 : 0);
      idxsm[row] = ix;
      idx_out[n0 + row] = (float)ix;
    }
  }
  __syncthreads();

  float lsum = 0.f;
  if (tid < ROWS_PER_BLOCK) {
    const int n   = n0 + tid;
    const int b2  = n >> 10;
    const int hw2 = n & (HWSZ - 1);
    const float* zb   = ze + (b2 << 18) + hw2;
    float*       qb   = zq + (b2 << 18) + hw2;
    const float* crow = cb + idxsm[tid] * KDIM;
#pragma unroll 4
    for (int c = 0; c < KDIM; ++c) {
      float q = crow[c];
      float x = zb[c << 10];
      float d = x - q;
      lsum += d * d;
      qb[c << 10] = q;
    }
  }
  red[tid] = lsum;
  __syncthreads();
  for (int s = 128; s > 0; s >>= 1) {
    if (tid < s) red[tid] += red[tid + s];
    __syncthreads();
  }
  if (tid == 0) loss_part[blockIdx.x] = red[0];
}

// ---------------------------------------------------------------------------
// Deterministic loss reduction. vq_loss = 1.25 * mean((ze - zq)^2)
// ---------------------------------------------------------------------------
__global__ void vq_loss_kernel(const float* __restrict__ part,
                               float* __restrict__ out) {
  __shared__ float red[128];
  red[threadIdx.x] = part[threadIdx.x];
  __syncthreads();
  for (int k = 64; k > 0; k >>= 1) {
    if (threadIdx.x < k) red[threadIdx.x] += red[threadIdx.x + k];
    __syncthreads();
  }
  if (threadIdx.x == 0) out[0] = red[0] * (1.25f / 4194304.0f);
}

// ---------------------------------------------------------------------------
extern "C" void kernel_launch(void* const* d_in, const int* in_sizes, int n_in,
                              void* d_out, int out_size, void* d_ws,
                              size_t ws_size, hipStream_t stream) {
  (void)in_sizes; (void)n_in; (void)out_size;
  const float* ze = (const float*)d_in[0];   // [16,256,32,32]
  const float* cb = (const float*)d_in[1];   // [8192,256]

  float* out  = (float*)d_out;
  float* loss = out;                 // [1]
  float* zqo  = out + 1;             // [4194304] BCHW
  float* idxf = out + 1 + 4194304;   // [16384] indices as float

  float* bias = (float*)d_ws;                       // [8192]
  float* part = bias + VOCAB;                       // [128]
  const size_t fixed = (size_t)(VOCAB + 128) * 4;   // 33280 B (16B aligned)
  unsigned short* cbhi = (unsigned short*)((char*)d_ws + fixed);
  unsigned short* cblo = cbhi + (size_t)VOCAB * KDIM;
  const size_t need = fixed + 2ull * VOCAB * KDIM * 2;  // 8,421,888 B

  if (ws_size >= need) {
    vq_prep_kernel<<<VOCAB / 8, 256, 0, stream>>>(cb, bias, cbhi, cblo);
    vq_main_async<<<N_TOK / ROWS_PER_BLOCK, 256, 0, stream>>>(
        ze, cb, cbhi, cblo, bias, part, zqo, idxf);
  } else {
    vq_bias_kernel<<<VOCAB / 8, 256, 0, stream>>>(cb, bias);
    vq_main_kernel<<<N_TOK / ROWS_PER_BLOCK, 256, 0, stream>>>(ze, cb, bias,
                                                               part, zqo, idxf);
  }
  vq_loss_kernel<<<1, 128, 0, stream>>>(part, loss);
}
